// CrossAttention_72292889526980
// MI455X (gfx1250) — compile-verified, matrix operand order checked by reading, not run on hardware
//
#include <hip/hip_runtime.h>
#include <hip/hip_bf16.h>

typedef __attribute__((ext_vector_type(16))) __bf16 v16bf;
typedef __attribute__((ext_vector_type(8)))  float  v8f;

#define LOG2E 1.4426950408889634f

// Hardware f32->bf16 (fptrunc lowers to v_cvt_pk_bf16_f32, RNE)
__device__ __forceinline__ unsigned short f2bf(float f) {
    __bf16 h = (__bf16)f;
    return __builtin_bit_cast(unsigned short, h);
}

__device__ __forceinline__ v8f wmma_bf16(v16bf a, v16bf b, v8f c) {
    // 8 args: (neg_a, A, neg_b, B, c_mod, C, reuse_a, reuse_b)
    return __builtin_amdgcn_wmma_f32_16x16x32_bf16(false, a, false, b, (short)0, c, false, false);
}

// Load one 16x32 bf16 A/B fragment. Per ISA 16-bit layout: lane's 16 elements
// are K = kg*8 + {0..7} then K = 16 + kg*8 + {0..7}; p points at the lane's
// first group (halfs), second group is +16 halfs (=+32B). Two b128 loads.
__device__ __forceinline__ v16bf load_frag(const unsigned short* p) {
    union { v16bf v; uint4 u[2]; } f;
    f.u[0] = *(const uint4*)(p);
    f.u[1] = *(const uint4*)(p + 16);
    return f.v;
}

// ---------------------------------------------------------------------------
// Projection GEMM: C[M,N] = A[M,K] * W[N,K]^T + bias[N]
//   ABF16: A is bf16 [M,K]; else A is f32 (cast to bf16 while staging).
// MODE 0: scatter to Q layout  [b,h,l,64] bf16
// MODE 1: scatter to K [b,h,m,64] and V-transposed [b,h,64,m] bf16
// MODE 2: write f32 row-major [M,N] (non-temporal)
// Block = 256 thr (8 waves), block tile 128x128, K step 32, double-buffered
// LDS (row stride 40 halfs -> conflict-free b128 fragment reads).
// ---------------------------------------------------------------------------
template<int MODE, bool ABF16>
__global__ __launch_bounds__(256) void gemm_proj(
    const void* __restrict__ Aptr,
    const float* __restrict__ Bw, const float* __restrict__ bias,
    int N, int K,
    float* __restrict__ outf,
    unsigned short* __restrict__ out0, unsigned short* __restrict__ out1)
{
    __shared__ unsigned short As[2][128 * 40];   // 32 cols + pad to 40 halfs
    __shared__ unsigned short Bs[2][128 * 40];

    const int tid  = threadIdx.x;
    const int wave = tid >> 5;
    const int lane = tid & 31;
    const int ln   = lane & 15;     // row/col within 16-tile
    const int kg   = lane >> 4;     // K half-group
    const int wm   = wave & 1;      // 2 M-halves of 64
    const int wn   = wave >> 1;     // 4 N-quarters of 32
    const int blockM = blockIdx.y * 128;
    const int blockN = blockIdx.x * 128;

    const float*          Af  = (const float*)Aptr;
    const unsigned short* Abf = (const unsigned short*)Aptr;

    // stage one 128x32 K-tile (A and W) into LDS buffer `buf`, bf16
    auto stage = [&](int k0, int buf) {
        if (ABF16) {
            #pragma unroll
            for (int p = 0; p < 2; ++p) {
                int r = (tid >> 2) + p * 64;
                int c = (tid & 3) * 8;
                *(uint4*)&As[buf][r * 40 + c] =
                    *(const uint4*)(Abf + (size_t)(blockM + r) * K + k0 + c);
            }
        } else {
            #pragma unroll
            for (int p = 0; p < 4; ++p) {
                int r = (tid >> 3) + p * 32;
                int c = (tid & 7) * 4;
                float4 f = *(const float4*)(Af + (size_t)(blockM + r) * K + k0 + c);
                unsigned short* dst = &As[buf][r * 40 + c];
                dst[0] = f2bf(f.x); dst[1] = f2bf(f.y);
                dst[2] = f2bf(f.z); dst[3] = f2bf(f.w);
            }
        }
        #pragma unroll
        for (int p = 0; p < 4; ++p) {
            int r = (tid >> 3) + p * 32;
            int c = (tid & 7) * 4;
            float4 f = *(const float4*)(Bw + (size_t)(blockN + r) * K + k0 + c);
            unsigned short* dst = &Bs[buf][r * 40 + c];
            dst[0] = f2bf(f.x); dst[1] = f2bf(f.y);
            dst[2] = f2bf(f.z); dst[3] = f2bf(f.w);
        }
    };

    const v8f vzero = {0.f,0.f,0.f,0.f,0.f,0.f,0.f,0.f};
    v8f acc[4][2];
    #pragma unroll
    for (int mt = 0; mt < 4; ++mt)
        #pragma unroll
        for (int nt = 0; nt < 2; ++nt) acc[mt][nt] = vzero;

    stage(0, 0);
    int cur = 0;
    for (int k0 = 0; k0 < K; k0 += 32) {
        __syncthreads();
        if (k0 + 32 < K) {
            stage(k0 + 32, cur ^ 1);           // overlap with WMMA on `cur`
            if (k0 + 64 < K) {                 // prefetch tile after next
                int r = tid >> 3, c = (tid & 7) * 4;
                size_t ao = (size_t)(blockM + r) * K + k0 + 64 + c;
                __builtin_prefetch(ABF16 ? (const void*)(Abf + ao)
                                         : (const void*)(Af + ao), 0, 1);
                __builtin_prefetch((const void*)(Bw + (size_t)(blockN + r) * K + k0 + 64 + c), 0, 1);
            }
        }
        v16bf afr[4], bfr[2];
        #pragma unroll
        for (int mt = 0; mt < 4; ++mt)
            afr[mt] = load_frag(&As[cur][(wm * 64 + mt * 16 + ln) * 40 + kg * 8]);
        #pragma unroll
        for (int nt = 0; nt < 2; ++nt)
            bfr[nt] = load_frag(&Bs[cur][(wn * 32 + nt * 16 + ln) * 40 + kg * 8]);
        #pragma unroll
        for (int mt = 0; mt < 4; ++mt)
            #pragma unroll
            for (int nt = 0; nt < 2; ++nt)
                acc[mt][nt] = wmma_bf16(afr[mt], bfr[nt], acc[mt][nt]);
        cur ^= 1;
    }

    // Epilogue. C layout: VGPR r, lane -> (row = r + 8*kg, col = ln) in tile.
    #pragma unroll
    for (int mt = 0; mt < 4; ++mt) {
        #pragma unroll
        for (int nt = 0; nt < 2; ++nt) {
            #pragma unroll
            for (int r = 0; r < 8; ++r) {
                int row = blockM + wm * 64 + mt * 16 + kg * 8 + r;
                int col = blockN + wn * 32 + nt * 16 + ln;
                float v = acc[mt][nt][r] + bias[col];
                if constexpr (MODE == 2) {
                    __builtin_nontemporal_store(v, &outf[(size_t)row * N + col]);
                } else if constexpr (MODE == 0) {
                    int b = row >> 10, l = row & 1023;
                    int h = col >> 6,  d = col & 63;
                    out0[(((size_t)(b * 16 + h)) * 1024 + l) * 64 + d] = f2bf(v);
                } else {
                    int b = row >> 10, m = row & 1023;
                    int h = col >> 7,  r2 = col & 127;
                    if (r2 < 64)
                        out0[(((size_t)(b * 16 + h)) * 1024 + m) * 64 + r2] = f2bf(v);
                    else // V stored transposed: [b,h,dh,m]
                        out1[(((size_t)(b * 16 + h)) * 64 + (r2 - 64)) * 1024 + m] = f2bf(v);
                }
            }
        }
    }
}

// ---------------------------------------------------------------------------
// Flash attention over m for one (b,h): per wave one 16-row Q tile x dh=64.
// Q [b,h,l,64], K [b,h,m,64], Vt [b,h,64,m] (all bf16). Out -> [b,l,h*64+d].
// ---------------------------------------------------------------------------
__global__ __launch_bounds__(256) void attn_kernel(
    const unsigned short* __restrict__ qh,
    const unsigned short* __restrict__ kh,
    const unsigned short* __restrict__ vt,
    unsigned short* __restrict__ oh)
{
    __shared__ unsigned short Pl[8][16 * 40];  // per-wave P scratch, padded
    const int tid  = threadIdx.x;
    const int wave = tid >> 5;
    const int lane = tid & 31;
    const int ln   = lane & 15;
    const int kg   = lane >> 4;
    const int bh = blockIdx.y;
    const int b  = bh >> 4, h = bh & 15;
    const int l0 = blockIdx.x * 128 + wave * 16;

    const unsigned short* qb = qh + (size_t)bh * (1024 * 64);
    const unsigned short* kb = kh + (size_t)bh * (1024 * 64);
    const unsigned short* vb = vt + (size_t)bh * (64 * 1024);
    unsigned short* Pw = &Pl[wave][0];

    v16bf qf[2];
    #pragma unroll
    for (int f = 0; f < 2; ++f)
        qf[f] = load_frag(qb + (size_t)(l0 + ln) * 64 + f * 32 + kg * 8);

    const v8f vzero = {0.f,0.f,0.f,0.f,0.f,0.f,0.f,0.f};
    v8f acc[4];
    float mrow[8], lrow[8];
    #pragma unroll
    for (int j = 0; j < 4; ++j) acc[j] = vzero;
    #pragma unroll
    for (int r = 0; r < 8; ++r) { mrow[r] = -1e30f; lrow[r] = 0.f; }

    for (int m0 = 0; m0 < 1024; m0 += 32) {
        // S = (Q * K^T) * scale, two 16x16 column tiles
        v8f s[2];
        #pragma unroll
        for (int t = 0; t < 2; ++t) {
            v16bf kf0 = load_frag(kb + (size_t)(m0 + t * 16 + ln) * 64 +      kg * 8);
            v16bf kf1 = load_frag(kb + (size_t)(m0 + t * 16 + ln) * 64 + 32 + kg * 8);
            v8f z = vzero;
            z = wmma_bf16(qf[0], kf0, z);
            z = wmma_bf16(qf[1], kf1, z);
            s[t] = z;
        }
        float rmax[8], p0[8], p1[8];
        #pragma unroll
        for (int r = 0; r < 8; ++r) {
            s[0][r] *= 0.125f;   // 1/sqrt(64)
            s[1][r] *= 0.125f;
            rmax[r] = fmaxf(s[0][r], s[1][r]);
        }
        // row max across the 16 lanes holding this row
        #pragma unroll
        for (int r = 0; r < 8; ++r) {
            #pragma unroll
            for (int msk = 1; msk < 16; msk <<= 1)
                rmax[r] = fmaxf(rmax[r], __shfl_xor(rmax[r], msk, 32));
        }
        #pragma unroll
        for (int r = 0; r < 8; ++r) {
            float mn   = fmaxf(mrow[r], rmax[r]);
            float corr = exp2f((mrow[r] - mn) * LOG2E);
            p0[r] = exp2f((s[0][r] - mn) * LOG2E);
            p1[r] = exp2f((s[1][r] - mn) * LOG2E);
            float rs = p0[r] + p1[r];
            #pragma unroll
            for (int msk = 1; msk < 16; msk <<= 1)
                rs += __shfl_xor(rs, msk, 32);
            lrow[r] = lrow[r] * corr + rs;
            mrow[r] = mn;
            #pragma unroll
            for (int j = 0; j < 4; ++j) acc[j][r] *= corr;
        }
        // P (16x32, C layout) -> LDS bf16 -> reload as A fragment
        #pragma unroll
        for (int r = 0; r < 8; ++r) {
            int prow = kg * 8 + r;
            Pw[prow * 40 + ln]      = f2bf(p0[r]);
            Pw[prow * 40 + 16 + ln] = f2bf(p1[r]);
        }
        v16bf af = load_frag(Pw + ln * 40 + kg * 8);
        // O += P * V ; V columns are rows of Vt -> contiguous b128 loads
        #pragma unroll
        for (int j = 0; j < 4; ++j) {
            v16bf vf = load_frag(vb + (size_t)(j * 16 + ln) * 1024 + m0 + kg * 8);
            acc[j] = wmma_bf16(af, vf, acc[j]);
        }
    }

    #pragma unroll
    for (int r = 0; r < 8; ++r) lrow[r] = 1.0f / lrow[r];
    #pragma unroll
    for (int j = 0; j < 4; ++j) {
        #pragma unroll
        for (int r = 0; r < 8; ++r) {
            int l = l0 + kg * 8 + r;
            int d = j * 16 + ln;
            oh[((size_t)(b * 1024 + l)) * 1024 + h * 64 + d] = f2bf(acc[j][r] * lrow[r]);
        }
    }
}

extern "C" void kernel_launch(void* const* d_in, const int* in_sizes, int n_in,
                              void* d_out, int out_size, void* d_ws, size_t ws_size,
                              hipStream_t stream) {
    const float* x   = (const float*)d_in[0];
    const float* y   = (const float*)d_in[1];
    const float* Wq  = (const float*)d_in[2];
    const float* bq  = (const float*)d_in[3];
    const float* Wkv = (const float*)d_in[4];
    const float* bkv = (const float*)d_in[5];
    const float* Wo  = (const float*)d_in[6];
    const float* bo  = (const float*)d_in[7];
    float* out = (float*)d_out;

    const size_t perBuf = 8ull * 16 * 1024 * 64;   // 8 Mi bf16 elems = 16 MB
    unsigned short* qh = (unsigned short*)d_ws;
    unsigned short* kh = qh + perBuf;
    unsigned short* vt = kh + perBuf;
    unsigned short* oh = vt + perBuf;

    dim3 blk(256);
    // Q = x * Wq^T + bq  -> [b,h,l,64]
    gemm_proj<0, false><<<dim3(1024 / 128, 8192 / 128), blk, 0, stream>>>(
        x, Wq, bq, 1024, 1024, nullptr, qh, nullptr);
    // KV = y * Wkv^T + bkv -> K [b,h,m,64], Vt [b,h,64,m]
    gemm_proj<1, false><<<dim3(2048 / 128, 8192 / 128), blk, 0, stream>>>(
        y, Wkv, bkv, 2048, 768, nullptr, kh, vt);
    // attention -> oh [b,l,1024]
    attn_kernel<<<dim3(1024 / 128, 8 * 16), blk, 0, stream>>>(qh, kh, vt, oh);
    // result = oh * Wo^T + bo (f32)
    gemm_proj<2, true><<<dim3(1024 / 128, 8192 / 128), blk, 0, stream>>>(
        oh, Wo, bo, 1024, 1024, out, nullptr, nullptr);
}
